// FullDPM_42116449305132
// MI455X (gfx1250) — matrix operand order repaired and checked by reading, not compiled, not run on hardware
//
#include <hip/hip_runtime.h>
#include <hip/hip_bf16.h>
#include <math.h>

// ---------------------------------------------------------------------------
// CDNA5 / gfx1250 fused EGNN diffusion step.
// Heavy math (edge MLP 2nd layer, all node GEMMs) runs on
// v_wmma_f32_16x16x32_bf16 (wave32, f32 accumulate).
// All WMMA operands are fetched as 16-byte (b128) loads:
//  - A fragments: bf16 row-major, the ISA K-pattern is two contiguous 16B runs
//  - B fragments: weights pre-packed once per GEMM into per-lane order
// ---------------------------------------------------------------------------

#define NNODES 2048
#define LG     128
#define NGRAPH 16
#define HID    128
#define LAT    32
#define EE     32
#define FEATK  288   // LAT + 2*HID

typedef __attribute__((ext_vector_type(16))) __bf16 v16bf;
typedef __attribute__((ext_vector_type(8)))  float  v8f;

union FragBF { v16bf v; uint4 q[2]; unsigned int u[8]; unsigned short s[16]; };
union FragF  { v8f   v; float f[8]; };

__device__ __forceinline__ unsigned short f2bf(float x) {
    unsigned u = __float_as_uint(x);
    return (unsigned short)((u + 0x7fffu + ((u >> 16) & 1u)) >> 16);
}

// ---------------------------------------------------------------------------
// Setup: DDPM noising, time embedding, bf16 feature pack [H_noisy|cond|t_emb]
// ---------------------------------------------------------------------------
__global__ void k_setup(const float* H0, const float* X0, const float* cond,
                        const float* epsH, const float* epsX,
                        const int* mask, const int* batch, const int* tarr,
                        float* Xn, float* Hn, float* xA, unsigned short* feat) {
    int nid = blockIdx.x * blockDim.x + threadIdx.x;
    if (nid >= NNODES) return;
    int t = tarr[batch[nid]];
    float beta = 1e-4f + (0.02f - 1e-4f) * (float)t / 100.0f;
    float ab = 1.0f;
    for (int i = 0; i <= t; ++i) {
        float b = 1e-4f + (0.02f - 1e-4f) * (float)i / 100.0f;
        ab *= (1.0f - b);
    }
    float m  = (mask[nid] != 0) ? 1.0f : 0.0f;
    float sa = sqrtf(ab), sb = sqrtf(1.0f - ab);
    for (int ax = 0; ax < 3; ++ax) {
        float x0 = X0[nid * 3 + ax];
        float xn = (m > 0.0f) ? sa * x0 + sb * epsX[nid * 3 + ax] : x0;
        Xn[nid * 3 + ax] = xn;
        xA[nid * 3 + ax] = xn;
    }
    for (int j = 0; j < LAT; ++j) {
        float h0 = H0[nid * LAT + j];
        float hn = (m > 0.0f) ? sa * h0 + sb * epsH[nid * LAT + j] : h0;
        Hn[nid * LAT + j] = hn;
        feat[nid * FEATK + j] = f2bf(hn);
    }
    for (int j = 0; j < HID; ++j)
        feat[nid * FEATK + LAT + j] = f2bf(cond[nid * HID + j]);
    for (int i = 0; i < 64; ++i) {
        float fr = expf(-logf(10000.0f) * (float)i / 63.0f);
        float a  = beta * fr;
        feat[nid * FEATK + LAT + HID + i]      = f2bf(sinf(a));
        feat[nid * FEATK + LAT + HID + 64 + i] = f2bf(cosf(a));
    }
}

// ---------------------------------------------------------------------------
// Pack W_f32[K x Nc] into WMMA-B per-lane fragment order (bf16):
// out[((nt*(K/32)+kc)*32 + lane)*16 + i] = bf16(W[(kc*32+(lane>>4)*8+(i&7)+(i>>3)*16)*Nc
//                                               + nt*16 + (lane&15)])
// Each (frag, lane) is 16 contiguous ushorts -> two b128 loads in the GEMM.
// ---------------------------------------------------------------------------
__global__ void k_packB(const float* W, unsigned short* out, int K, int Nc) {
    int idx = blockIdx.x * 256 + threadIdx.x;           // K*Nc total
    int i = idx & 15;
    int lane = (idx >> 4) & 31;
    int frag = idx >> 9;                                // nt*(K/32)+kc
    int kchunks = K >> 5;
    int kc = frag % kchunks, nt = frag / kchunks;
    int k = kc * 32 + (lane >> 4) * 8 + (i & 7) + (i >> 3) * 16;
    int n = nt * 16 + (lane & 15);
    out[idx] = f2bf(W[(size_t)k * Nc + n]);
}

// ---------------------------------------------------------------------------
// Generic bf16 WMMA GEMM: out = act(A_bf16[MxK] @ Bpacked[KxNc] + bias (+res))
// 128 threads = 4 waves; each wave owns one 16x16 output tile.
// ---------------------------------------------------------------------------
__global__ void __launch_bounds__(128)
k_gemm(const unsigned short* Abf, const unsigned short* Bp, const float* bias,
       const float* resid, float* outF, unsigned short* outBf,
       int M, int K, int Nc, int relu) {
    int wave = blockIdx.x * 4 + (threadIdx.x >> 5);
    int lane = threadIdx.x & 31;
    int ntiles = Nc >> 4;
    int mt = wave / ntiles, nt = wave % ntiles;
    int m0 = mt << 4, n0 = nt << 4;
    int lm = lane & 15, lh = lane >> 4;
    int kchunks = K >> 5;

    const unsigned short* arow = Abf + (size_t)(m0 + lm) * K + lh * 8;
    const unsigned short* brow = Bp + ((size_t)(nt * kchunks) * 32 + lane) * 16;

    v8f acc = {};
    for (int kc = 0; kc < kchunks; ++kc) {
        FragBF a, b;
        a.q[0] = *(const uint4*)(arow + kc * 32);
        a.q[1] = *(const uint4*)(arow + kc * 32 + 16);
        b.q[0] = *(const uint4*)(brow + kc * 512);
        b.q[1] = *(const uint4*)(brow + kc * 512 + 8);
        acc = __builtin_amdgcn_wmma_f32_16x16x32_bf16(
            false, a.v, false, b.v, (short)0, acc, false, false);
    }
    FragF d; d.v = acc;
    int n = n0 + lm;
    float bv = bias ? bias[n] : 0.0f;
#pragma unroll
    for (int i = 0; i < 8; ++i) {
        int m = m0 + i + lh * 8;
        float v = d.f[i] + bv;
        if (resid) v += resid[(size_t)m * Nc + n];
        if (relu)  v = v > 0.0f ? v : 0.0f;
        if (outF)  outF[(size_t)m * Nc + n] = v;
        if (outBf) outBf[(size_t)m * Nc + n] = f2bf(v);
    }
}

// et[t][k] = edge_emb[t] @ W1e[:,k]   (2x128, one block)
__global__ void k_et(const float* edge_emb, const float* W1e, float* et) {
    int idx = threadIdx.x;            // 256
    int t = idx >> 7, k = idx & 127;
    float s = 0.0f;
    for (int j = 0; j < EE; ++j) s += edge_emb[t * EE + j] * W1e[j * HID + k];
    et[idx] = s;
}

// cc = [h_bf | bf16(agg)]  (Nx256)
__global__ void k_concat(const unsigned short* hbf, const float* agg,
                         unsigned short* cc) {
    int idx = blockIdx.x * 256 + threadIdx.x;
    int nid = idx >> 8, j = idx & 255;
    cc[idx] = (j < HID) ? hbf[nid * HID + j] : f2bf(agg[nid * HID + (j - HID)]);
}

// ---------------------------------------------------------------------------
// Fused edge kernel. One workgroup = (graph g, 16-row tile rt).
// For 16 chunks of 8 cols: build m1 = relu(hA[r]+hB[c]+et+d2*w1d+b1) in LDS
// (bf16), then m = relu(m1 @ W2 + b2) via WMMA (W2 held in registers, one
// n-tile per wave), fold into agg (column-sum), px coefficients (shuffle
// reduce + LDS atomics) and the coordinate delta. No global atomics.
// ---------------------------------------------------------------------------
__global__ void __launch_bounds__(256)
k_edge(const float* hA, const float* hB, const float* x_in,
       const float* et, const float* b1, const float* W1d,
       const unsigned short* w2p, const float* b2,
       const float* pxW, const float* pxb,
       float* agg, float* x_out) {
    __shared__ unsigned short sm_m1[128 * 128];   // 32 KB edge-tile activations
    __shared__ float sm_hA[16 * 128];
    __shared__ float sm_hBc[8 * 128];
    __shared__ float sm_xg[128 * 3];
    __shared__ float sm_d2[128];
    __shared__ float sm_et[2 * 128];
    __shared__ float sm_b1[128], sm_w1d[128], sm_pxw[128], sm_b2[128];
    __shared__ float sm_coef[128];
    __shared__ float sm_xdel[16 * 3];

    int tid = threadIdx.x;
    int g = blockIdx.x >> 3, rt = blockIdx.x & 7;
    int nbase = g * LG;
    int rbase = nbase + rt * 16;
    int lane = tid & 31, w = tid >> 5;
    int lm = lane & 15, lh = lane >> 4;

    if (tid < 128) {
        sm_b1[tid]  = b1[tid];
        sm_w1d[tid] = W1d[tid];
        sm_pxw[tid] = pxW[tid];
        sm_b2[tid]  = b2[tid];
    }
    sm_et[tid] = et[tid];
    for (int i = tid; i < 128 * 3; i += 256) sm_xg[i] = x_in[nbase * 3 + i];
    for (int i = tid; i < 16 * 128; i += 256) sm_hA[i] = hA[(size_t)rbase * HID + i];
    if (tid < 48) sm_xdel[tid] = 0.0f;

    // B fragments: this wave's 16 output columns of W2, all 4 K-chunks.
    FragBF bfr[4];
#pragma unroll
    for (int kc = 0; kc < 4; ++kc) {
        const unsigned short* pb = w2p + (size_t)(((w * 4 + kc) * 32 + lane)) * 16;
        bfr[kc].q[0] = *(const uint4*)(pb);
        bfr[kc].q[1] = *(const uint4*)(pb + 8);
    }
    v8f aggacc = {};
    __syncthreads();
    float b2v  = sm_b2[w * 16 + lm];
    float pxwv = sm_pxw[w * 16 + lm];
    float pxbv = pxb[0];

    for (int chunk = 0; chunk < 16; ++chunk) {
        int cbase = chunk * 8;
        for (int i = tid; i < 8 * 128; i += 256)
            sm_hBc[i] = hB[(size_t)(nbase + cbase) * HID + i];
        if (tid < 128) {
            int e = tid, r = e & 15, c = cbase + (e >> 4);
            float dx = sm_xg[(rt * 16 + r) * 3 + 0] - sm_xg[c * 3 + 0];
            float dy = sm_xg[(rt * 16 + r) * 3 + 1] - sm_xg[c * 3 + 1];
            float dz = sm_xg[(rt * 16 + r) * 3 + 2] - sm_xg[c * 3 + 2];
            sm_d2[e]   = dx * dx + dy * dy + dz * dz;
            sm_coef[e] = pxbv;
        }
        __syncthreads();
        // m1 (factorized first edge layer) -> LDS, bf16; consecutive lanes
        // touch consecutive k (bank-conflict-free, broadcast-friendly).
#pragma unroll 4
        for (int v = 0; v < 64; ++v) {
            int idx = v * 256 + tid;
            int e = idx >> 7, k = idx & 127;
            int r = e & 15, cl = e >> 4;
            int rg = rt * 16 + r, cg = cbase + cl;
            int ty = ((rg >= 64) == (cg >= 64)) ? 0 : 1;
            float f = sm_hA[r * 128 + k] + sm_hBc[cl * 128 + k] +
                      sm_et[ty * 128 + k] + sm_d2[e] * sm_w1d[k] + sm_b1[k];
            f = f > 0.0f ? f : 0.0f;
            sm_m1[idx] = f2bf(f);
        }
        __syncthreads();
        // m = relu(m1 @ W2 + b2): 8 M-tiles x 4 K-chunks of WMMA per wave
        for (int mt = 0; mt < 8; ++mt) {
            v8f acc = {};
            const unsigned short* pa = sm_m1 + (mt * 16 + lm) * 128 + lh * 8;
#pragma unroll
            for (int kc = 0; kc < 4; ++kc) {
                FragBF a;
                a.q[0] = *(const uint4*)(pa + kc * 32);
                a.q[1] = *(const uint4*)(pa + kc * 32 + 16);
                acc = __builtin_amdgcn_wmma_f32_16x16x32_bf16(
                    false, a.v, false, bfr[kc].v, (short)0, acc, false, false);
            }
            FragF d; d.v = acc;
#pragma unroll
            for (int i = 0; i < 8; ++i) {
                float mv = d.f[i] + b2v;
                mv = mv > 0.0f ? mv : 0.0f;
                aggacc[i] += mv;            // segment-sum over cols (row-owned)
                d.f[i] = mv;
            }
            // coef[e] += sum_k m[e,k]*pxW[k]: reduce 16 lanes sharing an edge
#pragma unroll
            for (int i = 0; i < 8; ++i) {
                float v = d.f[i] * pxwv;
                v += __shfl_xor(v, 1, 32);
                v += __shfl_xor(v, 2, 32);
                v += __shfl_xor(v, 4, 32);
                v += __shfl_xor(v, 8, 32);
                if (lm == 0) atomicAdd(&sm_coef[mt * 16 + i + lh * 8], v);
            }
        }
        __syncthreads();
        if (tid < 128) {
            int e = tid, r = e & 15, c = cbase + (e >> 4);
            float cf = sm_coef[e];
#pragma unroll
            for (int ax = 0; ax < 3; ++ax) {
                float dv = sm_xg[(rt * 16 + r) * 3 + ax] - sm_xg[c * 3 + ax];
                atomicAdd(&sm_xdel[r * 3 + ax], dv * cf);
            }
        }
        __syncthreads();
    }
#pragma unroll
    for (int i = 0; i < 8; ++i)
        agg[(size_t)(rbase + i + lh * 8) * HID + w * 16 + lm] = aggacc[i];
    if (tid < 48) {
        int r = tid / 3, ax = tid % 3;
        x_out[(size_t)(rbase + r) * 3 + ax] =
            sm_xg[(rt * 16 + r) * 3 + ax] + sm_xdel[tid] * (1.0f / (float)LG);
    }
}

// ---------------------------------------------------------------------------
// Masked MSE losses -> d_out[0..1]
// ---------------------------------------------------------------------------
__global__ void k_final(const float* xf, const float* Xn, const float* epsX,
                        const float* hI, const float* Hn, const float* epsH,
                        const int* mask, float* out) {
    __shared__ float sX[256], sH[256], sC[256];
    int tid = threadIdx.x;
    float aX = 0, aH = 0, ac = 0;
    for (int nid = tid; nid < NNODES; nid += 256) {
        if (mask[nid] != 0) {
            ac += 1.0f;
            for (int ax = 0; ax < 3; ++ax) {
                float d = xf[nid * 3 + ax] - Xn[nid * 3 + ax] - epsX[nid * 3 + ax];
                aX += d * d;
            }
            for (int j = 0; j < LAT; ++j) {
                float d = hI[nid * LAT + j] - Hn[nid * LAT + j] - epsH[nid * LAT + j];
                aH += d * d;
            }
        }
    }
    sX[tid] = aX; sH[tid] = aH; sC[tid] = ac;
    __syncthreads();
    for (int s = 128; s > 0; s >>= 1) {
        if (tid < s) { sX[tid] += sX[tid + s]; sH[tid] += sH[tid + s]; sC[tid] += sC[tid + s]; }
        __syncthreads();
    }
    if (tid == 0) {
        float c = sC[0] + 1e-8f;
        out[0] = sX[0] / c;
        out[1] = sH[0] / c;
    }
}

// ---------------------------------------------------------------------------
extern "C" void kernel_launch(void* const* d_in, const int* in_sizes, int n_in,
                              void* d_out, int out_size, void* d_ws, size_t ws_size,
                              hipStream_t stream) {
    const float* H0     = (const float*)d_in[0];
    const float* X0     = (const float*)d_in[1];
    const float* cond   = (const float*)d_in[2];
    const float* epsH   = (const float*)d_in[3];
    const float* epsX   = (const float*)d_in[4];
    const float* in_W1  = (const float*)d_in[5];
    const float* in_b1  = (const float*)d_in[6];
    const float* in_W2  = (const float*)d_in[7];
    const float* in_b2  = (const float*)d_in[8];
    const float* in_W3  = (const float*)d_in[9];
    const float* in_b3  = (const float*)d_in[10];
    const float* edge_emb = (const float*)d_in[11];
    const float* h2i_W  = (const float*)d_in[12];
    const float* h2i_b  = (const float*)d_in[13];
    const float* pe_W1  = (const float*)d_in[14];
    const float* pe_b1  = (const float*)d_in[15];
    const float* pe_W2  = (const float*)d_in[16];
    const float* pe_b2  = (const float*)d_in[17];
    const float* ph_W1  = (const float*)d_in[18];
    const float* ph_b1  = (const float*)d_in[19];
    const float* ph_W2  = (const float*)d_in[20];
    const float* ph_b2  = (const float*)d_in[21];
    const float* px_W   = (const float*)d_in[22];
    const float* px_b   = (const float*)d_in[23];
    const int*   mask   = (const int*)d_in[24];
    const int*   batch  = (const int*)d_in[25];
    const int*   tarr   = (const int*)d_in[29];

    char* p = (char*)d_ws;
    auto allocF = [&](size_t n) { float* r = (float*)p; p += n * sizeof(float); return r; };
    auto allocU = [&](size_t n) { unsigned short* r = (unsigned short*)p; p += n * sizeof(unsigned short); return r; };

    float* Xn   = allocF(NNODES * 3);
    float* Hn   = allocF(NNODES * LAT);
    float* xA   = allocF(NNODES * 3);
    float* xB   = allocF(NNODES * 3);
    float* hF   = allocF(NNODES * HID);
    float* hAb  = allocF(NNODES * HID);
    float* hBb  = allocF(NNODES * HID);
    float* aggB = allocF(NNODES * HID);
    float* etB  = allocF(2 * HID);
    float* hI   = allocF(NNODES * LAT);
    unsigned short* feat = allocU(NNODES * FEATK);
    unsigned short* g1   = allocU(NNODES * HID);
    unsigned short* g2   = allocU(NNODES * HID);
    unsigned short* hbf  = allocU(NNODES * HID);
    unsigned short* ccB  = allocU(NNODES * 256);
    unsigned short* ubf  = allocU(NNODES * HID);
    unsigned short* wsB  = allocU(FEATK * HID);   // packed-B scratch (max K*Nc)

    k_setup<<<NNODES / 256, 256, 0, stream>>>(H0, X0, cond, epsH, epsX,
                                              mask, batch, tarr, Xn, Hn, xA, feat);

    const int GB128 = (NNODES / 16) * (HID / 16) / 4;   // 256 blocks
    auto packB = [&](const float* W, int K, int Nc) {
        k_packB<<<(K * Nc) / 256, 256, 0, stream>>>(W, wsB, K, Nc);
    };

    // input MLP
    packB(in_W1, FEATK, HID);
    k_gemm<<<GB128, 128, 0, stream>>>(feat, wsB, in_b1, nullptr, nullptr, g1,
                                      NNODES, FEATK, HID, 1);
    packB(in_W2, HID, HID);
    k_gemm<<<GB128, 128, 0, stream>>>(g1, wsB, in_b2, nullptr, nullptr, g2,
                                      NNODES, HID, HID, 1);
    packB(in_W3, HID, HID);
    k_gemm<<<GB128, 128, 0, stream>>>(g2, wsB, in_b3, nullptr, hF, hbf,
                                      NNODES, HID, HID, 0);

    float* xcur = xA;
    float* xnxt = xB;
    for (int l = 0; l < 3; ++l) {
        const float* W1l = pe_W1 + (size_t)l * 289 * HID;
        k_et<<<1, 256, 0, stream>>>(edge_emb, W1l + 256 * HID, etB);
        packB(W1l, HID, HID);
        k_gemm<<<GB128, 128, 0, stream>>>(hbf, wsB, nullptr, nullptr, hAb, nullptr,
                                          NNODES, HID, HID, 0);
        packB(W1l + 128 * HID, HID, HID);
        k_gemm<<<GB128, 128, 0, stream>>>(hbf, wsB, nullptr, nullptr, hBb, nullptr,
                                          NNODES, HID, HID, 0);
        packB(pe_W2 + (size_t)l * HID * HID, HID, HID);
        k_edge<<<NGRAPH * 8, 256, 0, stream>>>(hAb, hBb, xcur, etB,
                                               pe_b1 + l * HID, W1l + 288 * HID,
                                               wsB, pe_b2 + l * HID,
                                               px_W + l * HID, px_b + l,
                                               aggB, xnxt);
        k_concat<<<NNODES, 256, 0, stream>>>(hbf, aggB, ccB);
        packB(ph_W1 + (size_t)l * 256 * HID, 256, HID);
        k_gemm<<<GB128, 128, 0, stream>>>(ccB, wsB, ph_b1 + l * HID, nullptr,
                                          nullptr, ubf, NNODES, 256, HID, 1);
        packB(ph_W2 + (size_t)l * HID * HID, HID, HID);
        k_gemm<<<GB128, 128, 0, stream>>>(ubf, wsB, ph_b2 + l * HID, hF, hF, hbf,
                                          NNODES, HID, HID, 0);
        float* t = xcur; xcur = xnxt; xnxt = t;
    }
    // output head: hI = h @ h2i_W + h2i_b  (Nc = 32)
    packB(h2i_W, HID, LAT);
    k_gemm<<<(NNODES / 16) * (LAT / 16) / 4, 128, 0, stream>>>(
        hbf, wsB, h2i_b, nullptr, hI, nullptr, NNODES, HID, LAT, 0);

    k_final<<<1, 256, 0, stream>>>(xcur, Xn, epsX, hI, Hn, epsH, mask,
                                   (float*)d_out);
}
